// MatchSegmentation_37529424232899
// MI455X (gfx1250) — matrix operand order repaired and checked by reading, not compiled, not run on hardware
//
#include <hip/hip_runtime.h>
#include <hip/hip_bf16.h>

typedef __attribute__((ext_vector_type(16))) _Float16 v16h;
typedef __attribute__((ext_vector_type(8)))  _Float16 v8h;
typedef __attribute__((ext_vector_type(8)))  float    v8f;
typedef __attribute__((ext_vector_type(4)))  int      v4i;

#define SEG_EPS 1e-6f
#define WAVES_PER_BLOCK 8
#define MAX_BLOCKS 240
#define LDS_ROW 40            // 32 pixels + 8 halves pad; 80B rows keep b128 loads 16B aligned

// Convert 16 contiguous {0,1} int32s of one GT row into a f16 B-fragment slice.
__device__ __forceinline__ v16h load_gi_row16(const int* __restrict__ p) {
    v4i x0 = *(const v4i*)(p + 0);
    v4i x1 = *(const v4i*)(p + 4);
    v4i x2 = *(const v4i*)(p + 8);
    v4i x3 = *(const v4i*)(p + 12);
    v16h b;
#pragma unroll
    for (int i = 0; i < 4; ++i) {
        b[i]      = x0[i] ? (_Float16)1.0f : (_Float16)0.0f;
        b[4 + i]  = x1[i] ? (_Float16)1.0f : (_Float16)0.0f;
        b[8 + i]  = x2[i] ? (_Float16)1.0f : (_Float16)0.0f;
        b[12 + i] = x3[i] ? (_Float16)1.0f : (_Float16)0.0f;
    }
    return b;
}

// Coalesced seg load + log-diff + transpose-scatter into the wave-private LDS tile.
// linear idx = lane + 32*j -> pixel = idx / K, plane = idx % K   (valid for 16 <= K <= 32)
// KC != 0 : compile-time K (fully unrolled, immediate load offsets). KC == 0 : runtime K.
template <int KC>
__device__ __forceinline__ void scatter_diff(const float* __restrict__ sp,
                                             _Float16* __restrict__ dst,
                                             int lane, int Krt) {
    const int K = KC ? KC : Krt;
    int pixel = (lane >= K) ? 1 : 0;
    int plane = lane - pixel * K;
    if (KC) {
#pragma unroll
        for (int j = 0; j < (KC ? KC : 1); ++j) {
            float v = sp[lane + 32 * j];
            float d = __logf(v + SEG_EPS) - __logf(1.0f - v + SEG_EPS);
            dst[plane * LDS_ROW + pixel] = (_Float16)d;
            plane += 32 - KC; pixel += 1;          // advance idx by 32
            if (plane >= KC) { plane -= KC; pixel += 1; }
        }
    } else {
        for (int j = 0; j < K; ++j) {
            float v = sp[lane + 32 * j];
            float d = __logf(v + SEG_EPS) - __logf(1.0f - v + SEG_EPS);
            dst[plane * LDS_ROW + pixel] = (_Float16)d;
            plane += 32 - K; pixel += 1;
            if (plane >= K) { plane -= K; pixel += 1; }
        }
    }
}

// Phase 1: D[k,g] partial GEMM per block. Each wave owns a 32x32 output tile and
// walks 32-pixel chunks of the contraction dimension with v_wmma_f32_16x16x32_f16.
__global__ __launch_bounds__(256) void
phase1_wmma(const float* __restrict__ seg, const int* __restrict__ gt,
            const int* __restrict__ gnum, float* __restrict__ ws,
            int N, int K, int Gmax, int chunksPerWave, int totalChunks)
{
    __shared__ __align__(16) _Float16 sDiff[WAVES_PER_BLOCK][32 * LDS_ROW];
    __shared__ __align__(16) float    sAcc [WAVES_PER_BLOCK][1024];

    const int tid  = (int)threadIdx.x;
    const int wave = tid >> 5;
    const int lane = tid & 31;

    int G = *gnum;                       // scalar (s_load) broadcast
    if (G > Gmax) G = Gmax;
    if (G > 32)   G = 32;

    _Float16* myDiff = sDiff[wave];
    // Zero the wave-private diff tile once: rows >= K stay zero forever (A-tile-1 padding).
    {
        unsigned int* z = (unsigned int*)myDiff;
#pragma unroll
        for (int i = 0; i < (32 * LDS_ROW / 2) / 32; ++i)
            z[lane + 32 * i] = 0u;
    }

    v8f c00 = {}, c01 = {}, c10 = {}, c11 = {};

    // Force the per-wave chunk range into SGPRs: the chunk loop becomes pure SALU
    // control flow (EXEC provably all-ones at the WMMAs) and the seg base address
    // math is scalar, so the 21 seg loads use saddr + immediate-offset addressing.
    const int waveGlobal = __builtin_amdgcn_readfirstlane(
        (int)blockIdx.x * WAVES_PER_BLOCK + wave);
    const int chunk0 = waveGlobal * chunksPerWave;

    // A-fragment geometry (contiguous k = 16h + s order, same as B):
    //   lane<16: K = {kb..kb+15} with kb=0 ; lane>=16: kb=16
    const int mrow = lane & 15;
    const int kbA  = (lane < 16) ? 0 : 16;
    const _Float16* r0 = myDiff + mrow * LDS_ROW + kbA;
    const _Float16* r1 = myDiff + (mrow + 16) * LDS_ROW + kbA;
    // B-fragment geometry (contiguous k = 16h + s order, same as A): lanes 0-15 K=0..15, lanes 16-31 K=16..31
    const int kbB  = (lane < 16) ? 0 : 16;
    const int gcol = lane & 15;
    const bool hasB0 = (gcol < G);
    const bool hasB1 = (gcol + 16 < G);
    const int* gp0 = gt + (size_t)gcol * (size_t)N + kbB;          // row base, +n0 per chunk
    const int* gp1 = gt + (size_t)(gcol + 16) * (size_t)N + kbB;

    for (int c = 0; c < chunksPerWave; ++c) {
        const int chunk = chunk0 + c;
        if (chunk < totalChunks) {                  // scalar branch (chunk is in SGPR)
            const int n0 = chunk * 32;
            const float* sp = seg + (size_t)n0 * (size_t)K;

            if (chunk + 1 < totalChunks)            // speculative prefetch of next chunk
                __builtin_prefetch(sp + 32 * K, 0, 3);

            if (K == 21) scatter_diff<21>(sp, myDiff, lane, K);
            else         scatter_diff<0 >(sp, myDiff, lane, K);
            // same-wave LDS ops are in-order; compiler inserts the minimal s_wait_dscnt
            // before the dependent fragment loads below.

            // --- A fragments: two 16B ds_load_b128 each ---
            v8h a0lo = *(const v8h*)(r0);
            v8h a0hi = *(const v8h*)(r0 + 8);
            v8h a1lo = *(const v8h*)(r1);
            v8h a1hi = *(const v8h*)(r1 + 8);
            v16h a0 = __builtin_shufflevector(a0lo, a0hi, 0,1,2,3,4,5,6,7,8,9,10,11,12,13,14,15);
            v16h a1 = __builtin_shufflevector(a1lo, a1hi, 0,1,2,3,4,5,6,7,8,9,10,11,12,13,14,15);

            // --- B fragments: direct global b128 loads of GT rows ---
            v16h b0 = {}, b1 = {};
            if (hasB0) b0 = load_gi_row16(gp0 + n0);
            if (hasB1) b1 = load_gi_row16(gp1 + n0);

            // --- 32x32 tile MACC: D += diff * gi^T ---
            c00 = __builtin_amdgcn_wmma_f32_16x16x32_f16(false, a0, false, b0, (short)0, c00, false, false);
            c01 = __builtin_amdgcn_wmma_f32_16x16x32_f16(false, a0, false, b1, (short)0, c01, false, false);
            c10 = __builtin_amdgcn_wmma_f32_16x16x32_f16(false, a1, false, b0, (short)0, c10, false, false);
            c11 = __builtin_amdgcn_wmma_f32_16x16x32_f16(false, a1, false, b1, (short)0, c11, false, false);
        }
    }

    // --- spill C tiles to LDS using the documented f32 C/D layout ---
    // VGPR r, lane l : row m = (l<16 ? r : r+8), col = l%16
    {
        float* acc = sAcc[wave];
        const int rbase = (lane < 16) ? 0 : 8;
#pragma unroll
        for (int r = 0; r < 8; ++r) {
            const int m = rbase + r;
            acc[m * 32        + gcol     ] = c00[r];
            acc[m * 32        + gcol + 16] = c01[r];
            acc[(m + 16) * 32 + gcol     ] = c10[r];
            acc[(m + 16) * 32 + gcol + 16] = c11[r];
        }
    }
    __syncthreads();

    // deterministic fixed-order in-block reduction: 8 wave tiles -> 1 block partial in ws
    float* outp = ws + (size_t)blockIdx.x * 1024;
#pragma unroll
    for (int j = 0; j < 4; ++j) {
        const int e = j * 256 + tid;
        float s = 0.0f;
#pragma unroll
        for (int w = 0; w < WAVES_PER_BLOCK; ++w) s += sAcc[w][e];
        outp[e] = s;
    }
}

// Phase 2: fixed-order reduction of block partials + first-max argmax (== argmin of ce).
__global__ __launch_bounds__(256) void
phase2_argmax(const float* __restrict__ ws, const int* __restrict__ gnum,
              int* __restrict__ out, int K, int Gmax, int numBlocks)
{
    __shared__ float D[1024];
    const int tid = (int)threadIdx.x;
#pragma unroll
    for (int j = 0; j < 4; ++j) {
        const int e = j * 256 + tid;
        float s = 0.0f;
        for (int b = 0; b < numBlocks; ++b) s += ws[(size_t)b * 1024 + e];
        D[e] = s;
    }
    __syncthreads();

    int G = *gnum;
    if (G > Gmax) G = Gmax;
    if (G > 32)   G = 32;
    if (G < 1)    G = 1;

    if (tid < K) {
        float best = D[tid * 32];
        int   bi   = 0;
        for (int g = 1; g < G; ++g) {
            float v = D[tid * 32 + g];
            if (v > best) { best = v; bi = g; }   // strict '>' keeps first max == first min of ce
        }
        out[tid] = bi;
    }
}

extern "C" void kernel_launch(void* const* d_in, const int* in_sizes, int n_in,
                              void* d_out, int out_size, void* d_ws, size_t ws_size,
                              hipStream_t stream) {
    const float* seg  = (const float*)d_in[0];
    // d_in[1] = prob, unused by the reference result
    const int*   gt   = (const int*)d_in[2];
    const int*   gnum = (const int*)d_in[3];

    const int N    = in_sizes[1];            // prob is (N,1)
    const int K    = in_sizes[0] / N;        // 21 plane channels
    const int Gmax = in_sizes[2] / N;        // 21 GT rows present

    const int totalChunks = N / 32;          // N = 307200 is a multiple of 32

    // Clamp grid so ws (1024 f32 per block) can never overflow the scratch buffer.
    int nBlocks = (int)(ws_size / (1024 * sizeof(float)));
    if (nBlocks > MAX_BLOCKS) nBlocks = MAX_BLOCKS;
    if (nBlocks < 1)          nBlocks = 1;

    const int waves = nBlocks * WAVES_PER_BLOCK;
    const int chunksPerWave = (totalChunks + waves - 1) / waves;

    phase1_wmma<<<nBlocks, 256, 0, stream>>>(seg, gt, gnum, (float*)d_ws,
                                             N, K, Gmax, chunksPerWave, totalChunks);
    phase2_argmax<<<1, 256, 0, stream>>>((const float*)d_ws, gnum, (int*)d_out,
                                         K, Gmax, nBlocks);
}
